// ROIHead_31997506355978
// MI455X (gfx1250) — compile-verified, matrix operand order checked by reading, not run on hardware
//
#include <hip/hip_runtime.h>
#include <hip/hip_bf16.h>
#include <math.h>

// ---------------- types for WMMA / TDM (gfx1250, wave32) ----------------
typedef __attribute__((ext_vector_type(16))) _Float16 v16h;
typedef __attribute__((ext_vector_type(8)))  _Float16 v8h;
typedef __attribute__((ext_vector_type(8)))  float    v8f;
typedef __attribute__((ext_vector_type(4)))  unsigned int u32x4;
typedef __attribute__((ext_vector_type(8)))  int          i32x8;
typedef __attribute__((ext_vector_type(4)))  int          i32x4;

#if defined(__gfx1250__) && __has_builtin(__builtin_amdgcn_tensor_load_to_lds) && \
    __has_builtin(__builtin_amdgcn_s_wait_tensorcnt)
#define HAVE_TDM 1
#else
#define HAVE_TDM 0
#endif

// ---------------- problem constants ----------------
#define C_IN   512
#define POOL   7
#define FH     25
#define FW     25
#define NPROP  1000
#define MPAD   1024                 // proposals padded to WMMA multiple
#define K6     (C_IN * POOL * POOL) // 25088
#define HDIM   1024
#define NEGV   (-1.0e30f)
#define SCORE_TH 0.05f
#define NMS_TH   0.5f
#define TOPK     100
#define DW_CLAMP 4.1351666f         // log(1000/16)

// GEMM tiling
#define BM  128
#define BN  128
#define BK  32
#define LDT 40                      // padded LDS row stride in halfs (32 + 8)

// ================= ROI max pool -> f16 features =================
__global__ void roi_pool_kernel(const float* __restrict__ feat,
                                const float* __restrict__ props,
                                _Float16* __restrict__ X) {
  const int n = blockIdx.x;
  __shared__ int hs[POOL], he[POOL], wss[POOL], wee[POOL];
  if (threadIdx.x == 0) {
    float p0 = props[n * 4 + 0], p1 = props[n * 4 + 1];
    float p2 = props[n * 4 + 2], p3 = props[n * 4 + 3];
    int x1 = (int)floorf(p0 * 0.0625f + 0.5f);
    int y1 = (int)floorf(p1 * 0.0625f + 0.5f);
    int x2 = (int)floorf(p2 * 0.0625f + 0.5f);
    int y2 = (int)floorf(p3 * 0.0625f + 0.5f);
    float bw = (float)max(x2 - x1 + 1, 1) / (float)POOL;
    float bh = (float)max(y2 - y1 + 1, 1) / (float)POOL;
    for (int p = 0; p < POOL; ++p) {
      int a = (int)floorf((float)p * bh) + y1;
      int b = (int)ceilf((float)(p + 1) * bh) + y1;
      hs[p] = min(max(a, 0), FH);
      he[p] = min(max(b, 0), FH);
      a = (int)floorf((float)p * bw) + x1;
      b = (int)ceilf((float)(p + 1) * bw) + x1;
      wss[p] = min(max(a, 0), FW);
      wee[p] = min(max(b, 0), FW);
    }
  }
  __syncthreads();
  for (int idx = threadIdx.x; idx < C_IN * POOL * POOL; idx += blockDim.x) {
    int c = idx / (POOL * POOL);
    int cell = idx % (POOL * POOL);
    int py = cell / POOL, px = cell % POOL;
    const float* f = feat + (size_t)c * (FH * FW);
    float m = NEGV;
    for (int kh = 0; kh < 5; ++kh) {           // MAXBIN = 5
      int h = hs[py] + kh;
      if (h >= he[py]) break;
      int hh = min(h, FH - 1);
      for (int kw = 0; kw < 5; ++kw) {
        int w = wss[px] + kw;
        if (w >= wee[px]) break;
        int ww = min(w, FW - 1);
        m = fmaxf(m, f[hh * FW + ww]);
      }
    }
    float outv = (m > NEGV * 0.5f) ? m : 0.0f;
    X[(size_t)n * K6 + idx] = (_Float16)outv;
  }
}

// zero the padding rows of X (rows NPROP..MPAD-1)
__global__ void pad_rows_kernel(_Float16* __restrict__ X) {
  size_t i = (size_t)blockIdx.x * blockDim.x + threadIdx.x;
  size_t total = (size_t)(MPAD - NPROP) * K6;
  if (i < total) X[(size_t)NPROP * K6 + i] = (_Float16)0.0f;
}

// ================= TDM: 2D tile (BM rows x BK halfs) global -> padded LDS =================
#if HAVE_TDM
__device__ __forceinline__ void tdm_stage_a(const _Float16* gsrc, _Float16* ldst, int K) {
  // D# group 0: count=1 | lds_addr | global_addr(57b) | type=2
  unsigned long long ga = (unsigned long long)gsrc;
  unsigned lds_off = (unsigned)(unsigned long long)ldst;   // flat addr low 32 = LDS offset
  u32x4 g0;
  g0[0] = 1u;
  g0[1] = lds_off;
  g0[2] = (unsigned)ga;
  g0[3] = (unsigned)((ga >> 32) & 0x1FFFFFFu) | (2u << 30);
  // D# group 1: data_size=2B, pad_enable, pad_interval=16 DW (64B), pad_amount=4 DW (16B)
  // => LDS row stride = 64B + 16B = 80B = LDT halfs
  i32x8 g1;
  g1[0] = (int)((1u << 16) | (1u << 20) | (3u << 22) | (3u << 25));
  g1[1] = (int)(((unsigned)K & 0xFFFFu) << 16);                          // tensor_dim0 lo16
  g1[2] = (int)((((unsigned)K >> 16) & 0xFFFFu) | ((unsigned)MPAD << 16)); // dim0 hi | dim1 lo
  g1[3] = (int)((((unsigned)MPAD >> 16) & 0xFFFFu) | ((unsigned)BK << 16)); // dim1 hi | tile_dim0
  g1[4] = (int)(unsigned)BM;                                             // tile_dim1 (tile_dim2=0)
  g1[5] = K;                                                             // tensor_dim0_stride lo32
  g1[6] = 0;
  g1[7] = 0;
  i32x4 z4 = {0, 0, 0, 0};
  i32x8 z8 = {0, 0, 0, 0, 0, 0, 0, 0};
  // 6-arg toolchain variant: (group0, group1, group2, group3, extra_group, cpol)
  __builtin_amdgcn_tensor_load_to_lds(g0, g1, z4, z4, z8, 0);
}
#endif

// ================= WMMA GEMM: C = relu(A[f16] @ W[f32]^T + bias) =================
// Block: 256 threads (8 waves), 128x128 output tile, K-step 32, double-buffered LDS.
// Wave tile 32x64 (2 M-subtiles x 4 N-subtiles) -> 8 v_wmma per K-step.
// A tile staged by TDM (tensor_load_to_lds, hw padding -> LDT stride);
// W tile staged through VGPRs with inline fp32->f16 conversion.
__global__ __launch_bounds__(256)
void gemm_wmma_tiled_kernel(const _Float16* __restrict__ A,
                            const float* __restrict__ W,
                            const float* __restrict__ bias,
                            _Float16* __restrict__ C,
                            int N, int K, int do_relu) {
  __shared__ _Float16 As[2][BM * LDT];
  __shared__ _Float16 Bs[2][BM * LDT];

  const int tid  = threadIdx.x;
  const int lane = tid & 31;
  const int wave = tid >> 5;
  const int hseg = lane >> 4;      // K half-segment selector
  const int r    = lane & 15;
  const int wm   = wave & 3;       // wave M group: rows wm*32 .. +31
  const int wn   = wave >> 2;      // wave N group: cols wn*64 .. +63
  const int m0b  = blockIdx.y * BM;
  const int n0b  = blockIdx.x * BN;

  // W staging: thread stages 16 floats of one row-half per K-step
  const int wr_row = tid >> 1;          // 0..127 (row within N-panel)
  const int wr_kh  = (tid & 1) * 16;    // 0 or 16
  const float* wrow = W + (size_t)(n0b + wr_row) * K + wr_kh;
  const _Float16* apanel = A + (size_t)m0b * K;

  const int steps = K / BK;

  v8f acc[2][4];
#pragma unroll
  for (int a = 0; a < 2; ++a)
#pragma unroll
    for (int b = 0; b < 4; ++b) acc[a][b] = (v8f){0.f, 0.f, 0.f, 0.f, 0.f, 0.f, 0.f, 0.f};

  // ---- prologue: stage step 0 into buffer 0 ----
#if HAVE_TDM
  if (wave == 0) tdm_stage_a(apanel, &As[0][0], K);
#else
  {
    const _Float16* asrc = apanel + (size_t)wr_row * K + wr_kh;
    *(v8h*)(&As[0][wr_row * LDT + wr_kh])     = *(const v8h*)(asrc);
    *(v8h*)(&As[0][wr_row * LDT + wr_kh + 8]) = *(const v8h*)(asrc + 8);
  }
#endif
  {
    v8f w0 = *(const v8f*)(wrow);
    v8f w1 = *(const v8f*)(wrow + 8);
    *(v8h*)(&Bs[0][wr_row * LDT + wr_kh])     = __builtin_convertvector(w0, v8h);
    *(v8h*)(&Bs[0][wr_row * LDT + wr_kh + 8]) = __builtin_convertvector(w1, v8h);
  }
#if HAVE_TDM
  if (wave == 0) __builtin_amdgcn_s_wait_tensorcnt((short)0);
#endif
  __syncthreads();

  // ---- main loop: stage(kt+1) overlapped with compute(kt) ----
  for (int kt = 0; kt < steps; ++kt) {
    const int cur = kt & 1, nxt = cur ^ 1;
    const bool more = (kt + 1) < steps;

#if HAVE_TDM
    if (more && wave == 0)
      tdm_stage_a(apanel + (size_t)(kt + 1) * BK, &As[nxt][0], K);
#endif
    v8f w0, w1;
    if (more) {
      w0 = *(const v8f*)(wrow + (size_t)(kt + 1) * BK);
      w1 = *(const v8f*)(wrow + (size_t)(kt + 1) * BK + 8);
#if !HAVE_TDM
      const _Float16* asrc = apanel + (size_t)wr_row * K + (size_t)(kt + 1) * BK + wr_kh;
      *(v8h*)(&As[nxt][wr_row * LDT + wr_kh])     = *(const v8h*)(asrc);
      *(v8h*)(&As[nxt][wr_row * LDT + wr_kh + 8]) = *(const v8h*)(asrc + 8);
#endif
    }

    // compute on buffer `cur`
    v16h bfr[4];
#pragma unroll
    for (int ns = 0; ns < 4; ++ns) {
      const _Float16* bp = &Bs[cur][(wn * 64 + ns * 16 + r) * LDT + hseg * 8];
      v8h lo = *(const v8h*)bp;
      v8h hi = *(const v8h*)(bp + 16);
      bfr[ns] = __builtin_shufflevector(lo, hi,
          0, 1, 2, 3, 4, 5, 6, 7, 8, 9, 10, 11, 12, 13, 14, 15);
    }
#pragma unroll
    for (int ms = 0; ms < 2; ++ms) {
      const _Float16* ap = &As[cur][(wm * 32 + ms * 16 + r) * LDT + hseg * 8];
      v8h lo = *(const v8h*)ap;
      v8h hi = *(const v8h*)(ap + 16);
      v16h af = __builtin_shufflevector(lo, hi,
          0, 1, 2, 3, 4, 5, 6, 7, 8, 9, 10, 11, 12, 13, 14, 15);
#pragma unroll
      for (int ns = 0; ns < 4; ++ns)
        acc[ms][ns] = __builtin_amdgcn_wmma_f32_16x16x32_f16(
            false, af, false, bfr[ns], (short)0, acc[ms][ns], false, false);
    }

    if (more) {
      *(v8h*)(&Bs[nxt][wr_row * LDT + wr_kh])     = __builtin_convertvector(w0, v8h);
      *(v8h*)(&Bs[nxt][wr_row * LDT + wr_kh + 8]) = __builtin_convertvector(w1, v8h);
#if HAVE_TDM
      if (wave == 0) __builtin_amdgcn_s_wait_tensorcnt((short)0);
#endif
    }
    __syncthreads();
  }

  // ---- epilogue: bias + relu, store f16 ----
#pragma unroll
  for (int ns = 0; ns < 4; ++ns) {
    const int col = n0b + wn * 64 + ns * 16 + r;
    const float bv = bias[col];
#pragma unroll
    for (int ms = 0; ms < 2; ++ms) {
      const int rowb = m0b + wm * 32 + ms * 16 + 8 * hseg;
#pragma unroll
      for (int i = 0; i < 8; ++i) {
        float v = acc[ms][ns][i] + bv;
        if (do_relu) v = fmaxf(v, 0.0f);
        C[(size_t)(rowb + i) * N + col] = (_Float16)v;
      }
    }
  }
}

// ================= heads: cls(2) + reg(8), softmax, decode, clip =================
__global__ void heads_kernel(const _Float16* __restrict__ H7,
                             const float* __restrict__ wc, const float* __restrict__ bc,
                             const float* __restrict__ wr, const float* __restrict__ br,
                             const float* __restrict__ props,
                             const int* __restrict__ img_h, const int* __restrict__ img_w,
                             float* __restrict__ boxes,
                             float* __restrict__ score_m,
                             int* __restrict__ valid) {
  const int n = blockIdx.x;
  __shared__ float sh7[HDIM];
  __shared__ float part[10][128];
  for (int k = threadIdx.x; k < HDIM; k += blockDim.x)
    sh7[k] = (float)H7[(size_t)n * HDIM + k];
  __syncthreads();

  float acc[10];
#pragma unroll
  for (int j = 0; j < 10; ++j) acc[j] = 0.0f;
  for (int k = threadIdx.x; k < HDIM; k += blockDim.x) {
    float h = sh7[k];
    acc[0] += h * wc[k];
    acc[1] += h * wc[HDIM + k];
#pragma unroll
    for (int j = 0; j < 8; ++j) acc[2 + j] += h * wr[(size_t)j * HDIM + k];
  }
#pragma unroll
  for (int j = 0; j < 10; ++j) part[j][threadIdx.x] = acc[j];
  __syncthreads();

  if (threadIdx.x < 10) {
    float s = 0.0f;
    for (int t = 0; t < 128; ++t) s += part[threadIdx.x][t];
    s += (threadIdx.x < 2) ? bc[threadIdx.x] : br[threadIdx.x - 2];
    part[threadIdx.x][0] = s;
  }
  __syncthreads();

  if (threadIdx.x == 0) {
    float l0 = part[0][0], l1 = part[1][0];
    float mx = fmaxf(l0, l1);
    float e0 = __expf(l0 - mx), e1 = __expf(l1 - mx);
    float s1 = e1 / (e0 + e1);                   // softmax class-1 score
    float dx = part[6][0], dy = part[7][0];      // reg[1][0..3] = reg[4..7]
    float dw = fminf(part[8][0], DW_CLAMP);
    float dh = fminf(part[9][0], DW_CLAMP);
    float px1 = props[n * 4 + 0], py1 = props[n * 4 + 1];
    float px2 = props[n * 4 + 2], py2 = props[n * 4 + 3];
    float w = px2 - px1, h = py2 - py1;
    float cx = px1 + 0.5f * w, cy = py1 + 0.5f * h;
    float pcx = dx * w + cx, pcy = dy * h + cy;
    float pw = __expf(dw) * w, ph = __expf(dh) * h;
    float Wc = (float)(*img_w), Hc = (float)(*img_h);
    float bx1 = fminf(fmaxf(pcx - 0.5f * pw, 0.0f), Wc);
    float bx2 = fminf(fmaxf(pcx + 0.5f * pw, 0.0f), Wc);
    float by1 = fminf(fmaxf(pcy - 0.5f * ph, 0.0f), Hc);
    float by2 = fminf(fmaxf(pcy + 0.5f * ph, 0.0f), Hc);
    boxes[n * 4 + 0] = bx1; boxes[n * 4 + 1] = by1;
    boxes[n * 4 + 2] = bx2; boxes[n * 4 + 3] = by2;
    int v = (s1 > SCORE_TH) && ((bx2 - bx1) >= 1.0f) && ((by2 - by1) >= 1.0f);
    valid[n] = v;
    score_m[n] = v ? s1 : -1.0f;
  }
}

// ================= NMS + top-K (single 1024-thread block) =================
// Label offset in the reference is a uniform translation (all labels == 1 when
// NCLS==2) -> IoU-invariant -> omitted.
__global__ void nms_topk_kernel(const float* __restrict__ boxes,
                                const float* __restrict__ score_m,
                                const int* __restrict__ valid,
                                unsigned int* __restrict__ sup,   // [1024][32] in ws
                                float* __restrict__ out) {        // 400 + 100 + 100
  const int tid = threadIdx.x;
  __shared__ float ss[1024];
  __shared__ int   si[1024];
  __shared__ float sb[1024][4];
  __shared__ unsigned int skeep[32];
  __shared__ unsigned char korig[1024];

  ss[tid] = (tid < NPROP) ? score_m[tid] : -3.0e38f;
  si[tid] = tid;
  korig[tid] = 0;
  __syncthreads();

  // bitonic sort, descending by score
  for (int k = 2; k <= 1024; k <<= 1) {
    for (int j = k >> 1; j > 0; j >>= 1) {
      int ixj = tid ^ j;
      if (ixj > tid) {
        bool up = ((tid & k) == 0);
        bool swp = up ? (ss[tid] < ss[ixj]) : (ss[tid] > ss[ixj]);
        if (swp) {
          float t = ss[tid]; ss[tid] = ss[ixj]; ss[ixj] = t;
          int ti = si[tid]; si[tid] = si[ixj]; si[ixj] = ti;
        }
      }
      __syncthreads();
    }
  }

  // boxes in sorted order (positions >= NPROP hold padding, never referenced)
  {
    int o = si[tid];
    if (o < NPROP) {
      sb[tid][0] = boxes[o * 4 + 0];
      sb[tid][1] = boxes[o * 4 + 1];
      sb[tid][2] = boxes[o * 4 + 2];
      sb[tid][3] = boxes[o * 4 + 3];
    } else {
      sb[tid][0] = sb[tid][1] = sb[tid][2] = sb[tid][3] = 0.0f;
    }
  }
  __syncthreads();

  // suppression bit-matrix: sup[i][w] bit b set if IoU(i, w*32+b) > th, j > i
  for (int iw = tid; iw < 1024 * 32; iw += 1024) {
    int i = iw >> 5, w = iw & 31;
    unsigned int bits = 0u;
    if (i < NPROP) {
      float ax1 = sb[i][0], ay1 = sb[i][1], ax2 = sb[i][2], ay2 = sb[i][3];
      float aarea = (ax2 - ax1) * (ay2 - ay1);
      for (int b = 0; b < 32; ++b) {
        int j = w * 32 + b;
        if (j > i && j < NPROP) {
          float ix1 = fmaxf(ax1, sb[j][0]);
          float iy1 = fmaxf(ay1, sb[j][1]);
          float ix2 = fminf(ax2, sb[j][2]);
          float iy2 = fminf(ay2, sb[j][3]);
          float iww = fmaxf(ix2 - ix1, 0.0f);
          float ihh = fmaxf(iy2 - iy1, 0.0f);
          float inter = iww * ihh;
          float barea = (sb[j][2] - sb[j][0]) * (sb[j][3] - sb[j][1]);
          float iou = inter / fmaxf(aarea + barea - inter, 1e-6f);
          if (iou > NMS_TH) bits |= (1u << b);
        }
      }
    }
    sup[iw] = bits;
  }
  if (tid < 32) {
    unsigned int kw = 0u;
    for (int b = 0; b < 32; ++b) {
      int i = tid * 32 + b;
      if (i < NPROP) {
        int o = si[i];
        if (valid[o]) kw |= (1u << b);
      }
    }
    skeep[tid] = kw;
  }
  __syncthreads();

  if (tid == 0) {
    // greedy scan with 32-bit word masking
    for (int i = 0; i < NPROP; ++i) {
      if (skeep[i >> 5] & (1u << (i & 31))) {
        const unsigned int* srow = sup + (size_t)i * 32;
        for (int w = 0; w < 32; ++w) skeep[w] &= ~srow[w];
      }
    }
    // emit kept (sorted by score desc), then -1 fills by ascending orig index
    int cnt = 0;
    for (int i = 0; i < NPROP && cnt < TOPK; ++i) {
      if ((skeep[i >> 5] >> (i & 31)) & 1u) {
        int o = si[i];
        korig[o] = 1;
        out[cnt * 4 + 0] = sb[i][0];
        out[cnt * 4 + 1] = sb[i][1];
        out[cnt * 4 + 2] = sb[i][2];
        out[cnt * 4 + 3] = sb[i][3];
        out[4 * TOPK + cnt] = ss[i];
        out[5 * TOPK + cnt] = 1.0f;   // labels are all class 1
        ++cnt;
      }
    }
    for (int j = 0; j < NPROP && cnt < TOPK; ++j) {
      if (!korig[j]) {
        out[cnt * 4 + 0] = boxes[j * 4 + 0];
        out[cnt * 4 + 1] = boxes[j * 4 + 1];
        out[cnt * 4 + 2] = boxes[j * 4 + 2];
        out[cnt * 4 + 3] = boxes[j * 4 + 3];
        out[4 * TOPK + cnt] = -1.0f;
        out[5 * TOPK + cnt] = 1.0f;
        ++cnt;
      }
    }
  }
}

// ================= launch =================
extern "C" void kernel_launch(void* const* d_in, const int* in_sizes, int n_in,
                              void* d_out, int out_size, void* d_ws, size_t ws_size,
                              hipStream_t stream) {
  const float* feat  = (const float*)d_in[0];
  const float* props = (const float*)d_in[1];
  const float* w6    = (const float*)d_in[2];
  const float* b6    = (const float*)d_in[3];
  const float* w7    = (const float*)d_in[4];
  const float* b7    = (const float*)d_in[5];
  const float* wc    = (const float*)d_in[6];
  const float* bc    = (const float*)d_in[7];
  const float* wr    = (const float*)d_in[8];
  const float* br    = (const float*)d_in[9];
  const int* img_h   = (const int*)d_in[10];
  const int* img_w   = (const int*)d_in[11];
  float* out = (float*)d_out;

  // workspace carving
  char* w = (char*)d_ws;
  const size_t X_BYTES = (size_t)MPAD * K6 * sizeof(_Float16);     // 51,380,224
  const size_t H_BYTES = (size_t)MPAD * HDIM * sizeof(_Float16);   // 2,097,152
  _Float16* X  = (_Float16*)w;                       w += X_BYTES;
  _Float16* H6 = (_Float16*)w;                       w += H_BYTES;
  _Float16* H7 = (_Float16*)w;                       w += H_BYTES;
  float* boxes = (float*)w;                          w += 16384;   // 1000*4 f32
  float* score = (float*)w;                          w += 4096;    // 1000 f32
  int*   valid = (int*)w;                            w += 4096;    // 1000 i32
  unsigned int* sup = (unsigned int*)w;              w += 1024 * 32 * sizeof(unsigned int);

  // 1) ROI max pool -> f16 features, pad rows to MPAD
  roi_pool_kernel<<<NPROP, 256, 0, stream>>>(feat, props, X);
  {
    size_t total = (size_t)(MPAD - NPROP) * K6;
    pad_rows_kernel<<<(unsigned)((total + 255) / 256), 256, 0, stream>>>(X);
  }

  // 2) FC6: [1024 x 25088] @ w6^T -> [1024 x 1024], relu (WMMA, TDM-staged A)
  gemm_wmma_tiled_kernel<<<dim3(HDIM / BN, MPAD / BM), 256, 0, stream>>>(
      X, w6, b6, H6, HDIM, K6, 1);

  // 3) FC7: [1024 x 1024] @ w7^T -> [1024 x 1024], relu
  gemm_wmma_tiled_kernel<<<dim3(HDIM / BN, MPAD / BM), 256, 0, stream>>>(
      H6, w7, b7, H7, HDIM, HDIM, 1);

  // 4) heads + decode + clip + validity
  heads_kernel<<<NPROP, 128, 0, stream>>>(H7, wc, bc, wr, br, props,
                                          img_h, img_w, boxes, score, valid);

  // 5) NMS + top-100 emission
  nms_topk_kernel<<<1, 1024, 0, stream>>>(boxes, score, valid, sup, out);
}
